// _CLIPAttention_12790412607493
// MI455X (gfx1250) — matrix operand order repaired and checked by reading, not compiled, not run on hardware
//
#include <hip/hip_runtime.h>

// ---------------------------------------------------------------------------
// CLIP multi-head attention for gfx1250 (MI455X), B=8 T=1024 C=1024 H=16 D=64.
// Compute-bound (~103 GFLOP vs ~100MB traffic) -> f16 WMMA everywhere,
// f32 accumulate. 320KB LDS per WGP lets BOTH the full 32x1024 f32 score
// stripe (128KB) AND a transposed copy of the head's V matrix (128KB) live
// in LDS simultaneously -- no flash-attention bookkeeping, and all P*V
// B-fragments become contiguous 32B LDS reads.
// ---------------------------------------------------------------------------

typedef __attribute__((ext_vector_type(16))) _Float16 v16h;
typedef __attribute__((ext_vector_type(8)))  _Float16 v8h;
typedef __attribute__((ext_vector_type(8)))  float    v8f;

#define BDIM  8
#define TSEQ  1024
#define CDIM  1024
#define NHEAD 16
#define DHEAD 64
#define MTOT  (BDIM * TSEQ)   /* 8192 rows of x */

__device__ __forceinline__ v8f wmma16(v16h a, v16h b, v8f c) {
  // D = A(16x32 f16) * B(32x16 f16) + C(16x16 f32)
  return __builtin_amdgcn_wmma_f32_16x16x32_f16(
      /*neg_a=*/false, a, /*neg_b=*/false, b,
      /*c_mod=*/(short)0, c, /*reuse_a=*/false, /*reuse_b=*/false);
}

// A-fragment (16x32 f16, row-major source): lane (half, r) holds row r,
// K = half*16 + e (e<16)  -> two 16B chunks.
__device__ __forceinline__ v16h a_frag(const _Float16* __restrict__ row,
                                       int k0, int half) {
  v8h lo = *(const v8h*)(row + k0 + half * 16);
  v8h hi = *(const v8h*)(row + k0 + half * 16 + 8);
  v16h a;
#pragma unroll
  for (int i = 0; i < 8; ++i) { a[i] = lo[i]; a[8 + i] = hi[i]; }
  return a;
}

// ---------------------------------------------------------------------------
__global__ void k_f32_to_f16(const float* __restrict__ s,
                             _Float16* __restrict__ d, int n) {
  for (int i = blockIdx.x * blockDim.x + threadIdx.x; i < n;
       i += gridDim.x * blockDim.x)
    d[i] = (_Float16)s[i];
}

// ---------------------------------------------------------------------------
// Y = X * W^T + bias  (torch Linear). X:[8192,1024] f16, W:[1024,1024] f16.
// B-matrix fragment B[k][n] = W[n][k]: lane n reads 16 contiguous halves from
// W row n (same contiguous k order as a_frag: element e holds K = half*16 + e).
// MODE 0: store f16 into [B,H,T,D] head-split layout, scaled (QKV).
// MODE 1: store f32 row-major into d_out (output projection).
template <int MODE>
__global__ void k_gemm_xwT(const _Float16* __restrict__ X,
                           const _Float16* __restrict__ W,
                           const float* __restrict__ bias,
                           void* __restrict__ Ydst, float scale) {
  const int wave = threadIdx.x >> 5;
  const int lane = threadIdx.x & 31;
  const int r    = lane & 15;
  const int half = lane >> 4;
  const int row  = blockIdx.x * 128 + wave * 16 + r;   // A row for this lane
  const int colB = blockIdx.y * 64;

  v8f acc[4] = {};
  const _Float16* xrow = X + (size_t)row * CDIM;

  for (int k0 = 0; k0 < CDIM; k0 += 32) {
    v16h a = a_frag(xrow, k0, half);
    __builtin_prefetch(xrow + k0 + 512, 0, 0);  // global_prefetch_b8
    v16h b0 = *(const v16h*)(W + (size_t)(colB +  0 + r) * CDIM + k0 + half * 16);
    v16h b1 = *(const v16h*)(W + (size_t)(colB + 16 + r) * CDIM + k0 + half * 16);
    v16h b2 = *(const v16h*)(W + (size_t)(colB + 32 + r) * CDIM + k0 + half * 16);
    v16h b3 = *(const v16h*)(W + (size_t)(colB + 48 + r) * CDIM + k0 + half * 16);
    acc[0] = wmma16(a, b0, acc[0]);
    acc[1] = wmma16(a, b1, acc[1]);
    acc[2] = wmma16(a, b2, acc[2]);
    acc[3] = wmma16(a, b3, acc[3]);
  }

  // C/D layout: lane (half,n) holds N = n, M = half*8 + i in VGPR i.
  const int mBase = blockIdx.x * 128 + wave * 16 + half * 8;
#pragma unroll
  for (int j = 0; j < 4; ++j) {
    const int n  = colB + j * 16 + r;
    const float bv = bias[n];
    if (MODE == 0) {
      const int h = n >> 6, d = n & 63;
      _Float16* Y = (_Float16*)Ydst;
#pragma unroll
      for (int i = 0; i < 8; ++i) {
        const int m = mBase + i;
        const int bb = m >> 10, t = m & 1023;           // m = b*T + t
        Y[(((size_t)(bb * NHEAD + h) * TSEQ + t) << 6) + d] =
            (_Float16)((acc[j][i] + bv) * scale);
      }
    } else {
      float* Y = (float*)Ydst;
#pragma unroll
      for (int i = 0; i < 8; ++i) {
        const int m = mBase + i;
        Y[(size_t)m * CDIM + n] = acc[j][i] + bv;
      }
    }
  }
}

// ---------------------------------------------------------------------------
// Attention core. Grid: (T/32, B*H). Block: 256 threads (8 waves).
// Dynamic LDS layout:
//   S   : 32*1024 f32   (128 KB)  score stripe / softmax probabilities
//   redM: 256 f32, redS: 256 f32  (2 KB) softmax reductions
//   VT  : 64*1024 f16   (128 KB)  transposed V for this (b,h) head
// Total 258 KB (< 320 KB WGP LDS). Q already carries the 1/sqrt(D) scale.
__global__ void k_attention(const _Float16* __restrict__ Q,
                            const _Float16* __restrict__ K,
                            const _Float16* __restrict__ V,
                            _Float16* __restrict__ O) {
  extern __shared__ float smem[];
  float*    S    = smem;                       // 32 x 1024 f32
  float*    redM = smem + 32 * TSEQ;           // 256 f32
  float*    redS = redM + 256;                 // 256 f32
  _Float16* VT   = (_Float16*)(redS + 256);    // 64 x 1024 f16 (transposed V)

  const int bh   = blockIdx.y;        // b*16 + h
  const int q0   = blockIdx.x * 32;   // first query row of this stripe
  const int wave = threadIdx.x >> 5;
  const int lane = threadIdx.x & 31;
  const int r    = lane & 15;
  const int half = lane >> 4;

  const _Float16* Qp = Q + (size_t)bh * TSEQ * DHEAD + (size_t)q0 * DHEAD;
  const _Float16* Kp = K + (size_t)bh * TSEQ * DHEAD;
  const _Float16* Vp = V + (size_t)bh * TSEQ * DHEAD;

  // ---- Phase 0: stage V transposed into LDS: VT[d][t] = V[t][d] ----------
  // 65536 halves = 8192 coalesced v8h (16B) chunks; 32 chunks per thread.
  {
#pragma unroll 4
    for (int i = 0; i < 32; ++i) {
      const int cid  = i * 256 + threadIdx.x;    // chunk id
      const int base = cid * 8;                  // half index into V
      const int t    = base >> 6;                // row (time step)
      const int d0   = base & 63;                // first of 8 consecutive d
      v8h v = *(const v8h*)(Vp + base);
#pragma unroll
      for (int j = 0; j < 8; ++j)
        VT[(size_t)(d0 + j) * TSEQ + t] = v[j];
    }
  }

  // ---- Phase 1: S[32][1024] = Q_stripe * K^T  (K rows are B-fragments) ----
  {
    const int mt      = wave >> 2;    // 0..1 : 16-row strip
    const int quarter = wave & 3;     // 0..3 : 16 of the 64 N-tiles
    const _Float16* qrow = Qp + (size_t)(mt * 16 + r) * DHEAD;
    v16h a0 = a_frag(qrow, 0, half);
    v16h a1 = a_frag(qrow, 32, half);
    for (int nt = quarter * 16; nt < quarter * 16 + 16; ++nt) {
      const _Float16* krow = Kp + (size_t)(nt * 16 + r) * DHEAD;
      v16h b0 = *(const v16h*)(krow + 0  + half * 16);
      v16h b1 = *(const v16h*)(krow + 32 + half * 16);
      v8f c = {};
      c = wmma16(a0, b0, c);
      c = wmma16(a1, b1, c);
      float* srow = S + (size_t)(mt * 16 + half * 8) * TSEQ + nt * 16 + r;
#pragma unroll
      for (int i = 0; i < 8; ++i) srow[(size_t)i * TSEQ] = c[i];
    }
  }
  __syncthreads();

  // ---- Phase 2: row-wise softmax in LDS (8 threads per row) ----
  {
    const int row = threadIdx.x >> 3;   // 0..31
    const int seg = threadIdx.x & 7;    // 0..7 -> 128 cols each
    float* p = S + (size_t)row * TSEQ + seg * 128;
    float m = -3.0e38f;
#pragma unroll 4
    for (int i = 0; i < 128; ++i) m = fmaxf(m, p[i]);
    redM[threadIdx.x] = m;
    __syncthreads();
    float rm = redM[row * 8];
#pragma unroll
    for (int s2 = 1; s2 < 8; ++s2) rm = fmaxf(rm, redM[row * 8 + s2]);
    float sum = 0.f;
#pragma unroll 4
    for (int i = 0; i < 128; ++i) {
      float e = __expf(p[i] - rm);
      p[i] = e;
      sum += e;
    }
    redS[threadIdx.x] = sum;
    __syncthreads();
    float rs = 0.f;
#pragma unroll
    for (int s2 = 0; s2 < 8; ++s2) rs += redS[row * 8 + s2];
    const float rinv = 1.0f / rs;
#pragma unroll 4
    for (int i = 0; i < 128; ++i) p[i] *= rinv;
  }
  __syncthreads();

  // ---- Phase 3: O_stripe[32][64] = P[32][1024] * V[1024][64] ----
  // B[k][n] = V[k][n] = VT[n][k]: lane (half,n) reads 16 contiguous halves
  // from VT row n -> two ds_load_b128 instead of 16 strided global gathers.
  {
    const int mt     = wave >> 2;     // 0..1
    const int ntile  = wave & 3;      // 0..3
    const int rowIdx = mt * 16 + r;
    const int d      = ntile * 16 + r;
    const _Float16* vtrow = VT + (size_t)d * TSEQ;
    v8f c = {};
    for (int k0 = 0; k0 < TSEQ; k0 += 32) {
      v16h a;  // P fragment: gather f32 from LDS, convert to f16
#pragma unroll
      for (int e = 0; e < 16; ++e) {
        const int kk = k0 + half * 16 + e;
        a[e] = (_Float16)S[(size_t)rowIdx * TSEQ + kk];
      }
      v16h bf = *(const v16h*)(vtrow + k0 + half * 16);
      c = wmma16(a, bf, c);
    }
    const int bidx = bh >> 4;
    const int hh   = bh & 15;
    const int n    = hh * DHEAD + ntile * 16 + r;
#pragma unroll
    for (int i = 0; i < 8; ++i) {
      const int t = q0 + mt * 16 + half * 8 + i;
      O[(size_t)(bidx * TSEQ + t) * CDIM + n] = (_Float16)c[i];
    }
  }
}

// ---------------------------------------------------------------------------
extern "C" void kernel_launch(void* const* d_in, const int* in_sizes, int n_in,
                              void* d_out, int out_size, void* d_ws, size_t ws_size,
                              hipStream_t stream) {
  (void)in_sizes; (void)n_in; (void)out_size; (void)ws_size;
  const float* x  = (const float*)d_in[0];
  const float* wq = (const float*)d_in[1];
  const float* bq = (const float*)d_in[2];
  const float* wk = (const float*)d_in[3];
  const float* bk = (const float*)d_in[4];
  const float* wv = (const float*)d_in[5];
  const float* bv = (const float*)d_in[6];
  const float* wo = (const float*)d_in[7];
  const float* bo = (const float*)d_in[8];

  const size_t NX = (size_t)MTOT * CDIM;   // 8388608 elems
  const size_t NW = (size_t)CDIM * CDIM;   // 1048576 elems
  _Float16* Xh  = (_Float16*)d_ws;
  _Float16* Wqh = Xh  + NX;
  _Float16* Wkh = Wqh + NW;
  _Float16* Wvh = Wkh + NW;
  _Float16* Woh = Wvh + NW;
  _Float16* Qh  = Woh + NW;                // [B,H,T,D] f16
  _Float16* Kh  = Qh  + NX;
  _Float16* Vh  = Kh  + NX;
  _Float16* Oh  = Vh  + NX;                // [B,T,C] f16
  // total workspace: (5*NX + 4*NW)*2 bytes ~= 88 MB

  k_f32_to_f16<<<2048, 256, 0, stream>>>(x,  Xh,  (int)NX);
  k_f32_to_f16<<<512,  256, 0, stream>>>(wq, Wqh, (int)NW);
  k_f32_to_f16<<<512,  256, 0, stream>>>(wk, Wkh, (int)NW);
  k_f32_to_f16<<<512,  256, 0, stream>>>(wv, Wvh, (int)NW);
  k_f32_to_f16<<<512,  256, 0, stream>>>(wo, Woh, (int)NW);

  dim3 gg(MTOT / 128, CDIM / 64);
  // Q carries the attention scale D^-0.5 = 0.125 (applies to bias too: correct,
  // since the scale multiplies the whole q·k dot product).
  k_gemm_xwT<0><<<gg, 256, 0, stream>>>(Xh, Wqh, bq, (void*)Qh, 0.125f);
  k_gemm_xwT<0><<<gg, 256, 0, stream>>>(Xh, Wkh, bk, (void*)Kh, 1.0f);
  k_gemm_xwT<0><<<gg, 256, 0, stream>>>(Xh, Wvh, bv, (void*)Vh, 1.0f);

  // 128KB score stripe + 2KB reductions + 128KB transposed V = 258KB LDS
  const size_t smem_bytes =
      (size_t)32 * TSEQ * 4 + 2 * 256 * 4 + (size_t)DHEAD * TSEQ * 2;
  k_attention<<<dim3(TSEQ / 32, BDIM * NHEAD), 256, smem_bytes, stream>>>(Qh, Kh, Vh, Oh);

  k_gemm_xwT<1><<<gg, 256, 0, stream>>>(Oh, Woh, bo, d_out, 1.0f);
}
